// FastPointTransformer_81381040324813
// MI455X (gfx1250) — matrix operand order repaired and checked by reading, not compiled, not run on hardware
//
#include <hip/hip_runtime.h>
#include <hip/hip_bf16.h>
#include <math.h>
#include <stdint.h>

typedef __attribute__((ext_vector_type(16))) _Float16 v16h;
typedef __attribute__((ext_vector_type(8)))  _Float16 v8h;
typedef __attribute__((ext_vector_type(8)))  float    v8f;

__device__ __forceinline__ v8f wmma_f16(v16h a, v16h b, v8f c) {
  // D = A(16x32 f16) x B(32x16 f16) + C(16x16 f32)
  return __builtin_amdgcn_wmma_f32_16x16x32_f16(
      /*neg_a=*/false, a, /*neg_b=*/false, b,
      /*c_mod=*/(short)0, c, /*reuse_a=*/false, /*reuse_b=*/false);
}

// ---------------------------------------------------------------------------
// Tensor Data Mover: DMA a 64-row x 128-col f32 tile (contiguous, row-major)
// from global memory into LDS at byte offset `lds_off`.
// D# built in pinned, quad-aligned SGPRs (s[80:83] = group0, s[84:91] = group1,
// groups 2/3 = NULL -> 2D descriptor). Tracked by TENSORcnt.
//   group0: count=1 | lds_addr | global_addr[31:0] | global_addr[56:32],type=2
//   group1: data_size=4B | tensor_dim0=128 | tensor_dim1=64 | tile_dim0=128 |
//           tile_dim1=64 | tensor_dim0_stride=128
// ---------------------------------------------------------------------------
__device__ __forceinline__ void tdm_load_rows64(uint32_t lds_off, const float* gsrc) {
  uint64_t ga = (uint64_t)(uintptr_t)gsrc;
  uint32_t lo = (uint32_t)ga;
  uint32_t hi = (uint32_t)((ga >> 32) & 0x01FFFFFFu) | (2u << 30);
  asm volatile(
      "s_mov_b32 s80, 1\n\t"
      "s_mov_b32 s81, %0\n\t"
      "s_mov_b32 s82, %1\n\t"
      "s_mov_b32 s83, %2\n\t"
      "s_mov_b32 s84, 0x20000\n\t"    // data_size = 4 bytes
      "s_mov_b32 s85, 0x800000\n\t"   // tensor_dim0 = 128
      "s_mov_b32 s86, 0x400000\n\t"   // tensor_dim1 = 64
      "s_mov_b32 s87, 0x800000\n\t"   // tile_dim0 = 128
      "s_mov_b32 s88, 64\n\t"         // tile_dim1 = 64, tile_dim2 = 0
      "s_mov_b32 s89, 128\n\t"        // tensor_dim0_stride = 128
      "s_mov_b32 s90, 0\n\t"
      "s_mov_b32 s91, 0\n\t"
      "tensor_load_to_lds s[80:83], s[84:91]"
      :: "s"(lds_off), "s"(lo), "s"(hi)
      : "s80","s81","s82","s83","s84","s85","s86","s87",
        "s88","s89","s90","s91","memory");
}

// ---------------------------------------------------------------------------
// K0: zero the accumulation buffer (d_out is poisoned by harness)
// ---------------------------------------------------------------------------
__global__ __launch_bounds__(256) void zero_kernel(float4* __restrict__ p, int n4) {
  int i = blockIdx.x * 256 + threadIdx.x;
  if (i < n4) p[i] = make_float4(0.f, 0.f, 0.f, 0.f);
}

// ---------------------------------------------------------------------------
// K1: fused point MLP: t = relu(bn2(relu(bn1(p@w1))@w2))  -> f16 [N,128]
// ---------------------------------------------------------------------------
__global__ __launch_bounds__(256) void mlp_kernel(
    const float* __restrict__ pts, const float* __restrict__ w1,
    const float* __restrict__ s1, const float* __restrict__ sh1,
    const float* __restrict__ w2, const float* __restrict__ s2,
    const float* __restrict__ sh2, _Float16* __restrict__ t, int nrows) {
  int idx = blockIdx.x * 256 + threadIdx.x;
  if (idx >= nrows * 128) return;
  int pt = idx >> 7, c = idx & 127;
  float p0 = pts[pt * 3 + 0], p1 = pts[pt * 3 + 1], p2 = pts[pt * 3 + 2];
  float h[3];
#pragma unroll
  for (int j = 0; j < 3; j++) {
    float a = p0 * w1[0 * 3 + j] + p1 * w1[1 * 3 + j] + p2 * w1[2 * 3 + j];
    h[j] = fmaxf(a * s1[j] + sh1[j], 0.f);
  }
  float o = h[0] * w2[0 * 128 + c] + h[1] * w2[1 * 128 + c] + h[2] * w2[2 * 128 + c];
  o = fmaxf(o * s2[c] + sh2[c], 0.f);
  t[idx] = (_Float16)o;
}

// ---------------------------------------------------------------------------
// K2: x = feats + t @ w3 + b3  (WMMA) -> x_f16 [N,128]
// Weight staged via TDM (2 x 64-row DMA) then transpose-converted to f16 LDS.
// ---------------------------------------------------------------------------
__global__ __launch_bounds__(256) void gemm_x_kernel(
    const _Float16* __restrict__ t, const float* __restrict__ feats,
    const float* __restrict__ w3, const float* __restrict__ b3,
    _Float16* __restrict__ xh, int nrows) {
  __shared__ alignas(16) float stage[64 * 128];   // 32 KB TDM landing pad
  __shared__ alignas(32) _Float16 lw[128 * 128];  // lw[n*128 + k] = w3[k][n]
  const int tid = threadIdx.x;
  const uint32_t stage_off = (uint32_t)(uintptr_t)(void*)stage;
#pragma unroll
  for (int half = 0; half < 2; half++) {
    if (tid < 32) {                                // one wave issues the DMA
      tdm_load_rows64(stage_off, w3 + half * 64 * 128);
      __builtin_amdgcn_s_wait_tensorcnt(0);
    }
    __syncthreads();
    for (int i = tid; i < 64 * 128; i += 256) {
      const int k = (half << 6) + (i >> 7), nc = i & 127;
      lw[nc * 128 + k] = (_Float16)stage[i];
    }
    __syncthreads();
  }
  const int wave = tid >> 5, lane = tid & 31;
  const int rt = blockIdx.x * 8 + wave;
  if (rt * 16 >= nrows) return;                    // wave-uniform: EXEC all-1 for WMMA
  const int mrow = rt * 16 + (lane & 15);
  const int akb = (lane >> 4) * 16;                // 16-bit A layout: half-wave K split
  v16h afrag[4];
  const _Float16* tb = t + (size_t)mrow * 128;
#pragma unroll
  for (int ks = 0; ks < 4; ks++) {
    v8h lo = *(const v8h*)(tb + ks * 32 + akb);
    v8h hi = *(const v8h*)(tb + ks * 32 + akb + 8);
    afrag[ks] = __builtin_shufflevector(lo, hi, 0,1,2,3,4,5,6,7,8,9,10,11,12,13,14,15);
  }
  const int bkb = (lane >> 4) * 16;                // 16-bit B layout: half-wave K split
  const int nc = lane & 15;
  const int rbase = rt * 16 + (lane >> 4) * 8;     // C layout rows
#pragma unroll
  for (int ct = 0; ct < 8; ct++) {
    v8f acc = {};
    const _Float16* bp = &lw[(ct * 16 + nc) * 128 + bkb];
#pragma unroll
    for (int ks = 0; ks < 4; ks++)
      acc = wmma_f16(afrag[ks], *(const v16h*)(bp + ks * 32), acc);
    const int col = ct * 16 + nc;
    const float bias = b3[col];
    float fv[8];
#pragma unroll
    for (int i = 0; i < 8; i++)                    // one load clause, no interleave
      fv[i] = feats[(size_t)(rbase + i) * 128 + col];
#pragma unroll
    for (int i = 0; i < 8; i++)
      xh[(size_t)(rbase + i) * 128 + col] = (_Float16)(acc[i] + bias + fv[i]);
  }
}

// ---------------------------------------------------------------------------
// K3: q = l2norm_per_head(x@wq + bq) -> nq f32;  v = x@wv + bv -> v f32
// A fragments reused across both GEMMs; head == 16-col tile (CH=16).
// ---------------------------------------------------------------------------
__global__ __launch_bounds__(256) void gemm_qv_kernel(
    const _Float16* __restrict__ xh,
    const float* __restrict__ wq, const float* __restrict__ bq,
    const float* __restrict__ wv, const float* __restrict__ bv,
    float* __restrict__ nq, float* __restrict__ vout, int nrows) {
  __shared__ alignas(32) _Float16 lq[128 * 128];
  __shared__ alignas(32) _Float16 lv[128 * 128];
  const int tid = threadIdx.x;
  for (int i = tid; i < 128 * 128; i += 256) {
    const int k = i >> 7, nc = i & 127;
    lq[nc * 128 + k] = (_Float16)wq[i];
    lv[nc * 128 + k] = (_Float16)wv[i];
  }
  __syncthreads();
  const int wave = tid >> 5, lane = tid & 31;
  const int rt = blockIdx.x * 8 + wave;
  if (rt * 16 >= nrows) return;
  const int mrow = rt * 16 + (lane & 15);
  const int akb = (lane >> 4) * 16;
  v16h afrag[4];
  const _Float16* tb = xh + (size_t)mrow * 128;
#pragma unroll
  for (int ks = 0; ks < 4; ks++) {
    v8h lo = *(const v8h*)(tb + ks * 32 + akb);
    v8h hi = *(const v8h*)(tb + ks * 32 + akb + 8);
    afrag[ks] = __builtin_shufflevector(lo, hi, 0,1,2,3,4,5,6,7,8,9,10,11,12,13,14,15);
  }
  const int bkb = (lane >> 4) * 16;
  const int nc = lane & 15;
  const int rbase = rt * 16 + (lane >> 4) * 8;
#pragma unroll
  for (int ct = 0; ct < 8; ct++) {                 // Q + per-(row,head) L2 norm
    v8f acc = {};
    const _Float16* bp = &lq[(ct * 16 + nc) * 128 + bkb];
#pragma unroll
    for (int ks = 0; ks < 4; ks++)
      acc = wmma_f16(afrag[ks], *(const v16h*)(bp + ks * 32), acc);
    const int col = ct * 16 + nc;
    const float bias = bq[col];
#pragma unroll
    for (int i = 0; i < 8; i++) {
      float qv = acc[i] + bias;
      float s = qv * qv;                           // reduce over 16 lanes (half-wave)
      s += __shfl_xor(s, 1); s += __shfl_xor(s, 2);
      s += __shfl_xor(s, 4); s += __shfl_xor(s, 8);
      const float sc = 1.0f / fmaxf(sqrtf(s), 1e-12f);
      nq[(size_t)(rbase + i) * 128 + col] = qv * sc;
    }
  }
#pragma unroll
  for (int ct = 0; ct < 8; ct++) {                 // V
    v8f acc = {};
    const _Float16* bp = &lv[(ct * 16 + nc) * 128 + bkb];
#pragma unroll
    for (int ks = 0; ks < 4; ks++)
      acc = wmma_f16(afrag[ks], *(const v16h*)(bp + ks * 32), acc);
    const int col = ct * 16 + nc;
    const float bias = bv[col];
#pragma unroll
    for (int i = 0; i < 8; i++)
      vout[(size_t)(rbase + i) * 128 + col] = acc[i] + bias;
  }
}

// ---------------------------------------------------------------------------
// K4: npos = l2norm(pos_enc) per (k,head) over 16 channels  [27*128]
// ---------------------------------------------------------------------------
__global__ __launch_bounds__(256) void npos_kernel(
    const float* __restrict__ pe, float* __restrict__ npos, int total) {
  int i = blockIdx.x * 256 + threadIdx.x;
  if (i >= total) return;
  float x = pe[i];
  float s = x * x;
  s += __shfl_xor(s, 1); s += __shfl_xor(s, 2);
  s += __shfl_xor(s, 4); s += __shfl_xor(s, 8);
  npos[i] = x / fmaxf(sqrtf(s), 1e-12f);
}

// ---------------------------------------------------------------------------
// K5: attention gather->dot->scatter. 16 lanes per pair, one channel each.
// nq/v/out are L2-resident (30.7 MB each vs 192 MB L2).
// ---------------------------------------------------------------------------
__global__ __launch_bounds__(256) void attn_kernel(
    const long long* __restrict__ row0, const long long* __restrict__ row1,
    const float* __restrict__ nq, const float* __restrict__ npos,
    const float* __restrict__ v, float* __restrict__ aout, int m) {
  int g = (blockIdx.x * 256 + threadIdx.x) >> 4;   // pair index
  int c = threadIdx.x & 15;                        // channel within head
  if (g >= m) return;
  unsigned r0 = (unsigned)row0[g];                 // < N*K = 1.62M: 32-bit div magic
  int qi = (int)row1[g];
  unsigned koff = r0 % 27u;
  unsigned kidx = r0 / 27u;
  const float* nqp = nq + (size_t)qi * 128;
  const float* npp = npos + (size_t)koff * 128;
  const float* vp = v + (size_t)kidx * 128;
  float* op = aout + (size_t)qi * 128;
  __builtin_prefetch(nqp, 0, 1);                   // global_prefetch_b8 (L2-resident)
  __builtin_prefetch(vp, 0, 1);
#pragma unroll
  for (int h = 0; h < 8; h++) {
    float p = nqp[h * 16 + c] * npp[h * 16 + c];
    p += __shfl_xor(p, 1); p += __shfl_xor(p, 2);
    p += __shfl_xor(p, 4); p += __shfl_xor(p, 8);  // all 16 lanes hold the dot
    atomicAdd(op + h * 16 + c, p * vp[h * 16 + c]);
  }
}

// ---------------------------------------------------------------------------
// K6: in-place output projection: out = out @ wo + bo  (WMMA, TDM-staged W)
// Each wave reads its own 16 rows fully into A fragments before overwriting.
// ---------------------------------------------------------------------------
__global__ __launch_bounds__(256) void gemm_o_kernel(
    float* __restrict__ out, const float* __restrict__ wo,
    const float* __restrict__ bo, int nrows) {
  __shared__ alignas(16) float stage[64 * 128];
  __shared__ alignas(32) _Float16 lw[128 * 128];
  const int tid = threadIdx.x;
  const uint32_t stage_off = (uint32_t)(uintptr_t)(void*)stage;
#pragma unroll
  for (int half = 0; half < 2; half++) {
    if (tid < 32) {
      tdm_load_rows64(stage_off, wo + half * 64 * 128);
      __builtin_amdgcn_s_wait_tensorcnt(0);
    }
    __syncthreads();
    for (int i = tid; i < 64 * 128; i += 256) {
      const int k = (half << 6) + (i >> 7), nc = i & 127;
      lw[nc * 128 + k] = (_Float16)stage[i];
    }
    __syncthreads();
  }
  const int wave = tid >> 5, lane = tid & 31;
  const int rt = blockIdx.x * 8 + wave;
  if (rt * 16 >= nrows) return;
  const int mrow = rt * 16 + (lane & 15);
  const int akb = (lane >> 4) * 16;
  const float* ab = out + (size_t)mrow * 128;
  v16h afrag[4];
#pragma unroll
  for (int ks = 0; ks < 4; ks++) {
    v16h a;
#pragma unroll
    for (int j = 0; j < 8; j++) {
      a[j] = (_Float16)ab[ks * 32 + akb + j];
      a[j + 8] = (_Float16)ab[ks * 32 + akb + 8 + j];
    }
    afrag[ks] = a;
  }
  const int bkb = (lane >> 4) * 16;
  const int nc = lane & 15;
  const int rbase = rt * 16 + (lane >> 4) * 8;
#pragma unroll
  for (int ct = 0; ct < 8; ct++) {
    v8f acc = {};
    const _Float16* bp = &lw[(ct * 16 + nc) * 128 + bkb];
#pragma unroll
    for (int ks = 0; ks < 4; ks++)
      acc = wmma_f16(afrag[ks], *(const v16h*)(bp + ks * 32), acc);
    const int col = ct * 16 + nc;
    const float bias = bo[col];
#pragma unroll
    for (int i = 0; i < 8; i++)
      out[(size_t)(rbase + i) * 128 + col] = acc[i] + bias;
  }
}

// ---------------------------------------------------------------------------
extern "C" void kernel_launch(void* const* d_in, const int* in_sizes, int n_in,
                              void* d_out, int out_size, void* d_ws, size_t ws_size,
                              hipStream_t stream) {
  const float* feats = (const float*)d_in[0];
  const float* pts   = (const float*)d_in[1];
  const float* w1    = (const float*)d_in[2];
  const float* s1    = (const float*)d_in[3];
  const float* sh1   = (const float*)d_in[4];
  const float* w2    = (const float*)d_in[5];
  const float* s2    = (const float*)d_in[6];
  const float* sh2   = (const float*)d_in[7];
  const float* w3    = (const float*)d_in[8];
  const float* b3    = (const float*)d_in[9];
  const float* wq    = (const float*)d_in[10];
  const float* bq    = (const float*)d_in[11];
  const float* wv    = (const float*)d_in[12];
  const float* bv    = (const float*)d_in[13];
  const float* wo    = (const float*)d_in[14];
  const float* bo    = (const float*)d_in[15];
  const float* pe    = (const float*)d_in[16];
  const long long* kq0 = (const long long*)d_in[17];
  const long long* kq1 = (const long long*)d_in[18];

  const int N = in_sizes[0] / 128;
  const int M = in_sizes[17];
  float* out = (float*)d_out;

  // Workspace layout (bytes): t_f16 | x_f16 | nq_f32 | v_f32 | npos_f32
  char* ws = (char*)d_ws;
  _Float16* t_h   = (_Float16*)(ws);
  _Float16* x_h   = (_Float16*)(ws + (size_t)N * 256);
  float*    nqW   = (float*)(ws + (size_t)N * 512);
  float*    vW    = (float*)(ws + (size_t)N * 1024);
  float*    nposW = (float*)(ws + (size_t)N * 1536);

  const int n128 = N * 128;
  const int gemmBlocks = ((N + 15) / 16 + 7) / 8;   // 8 row-tiles of 16 per block

  zero_kernel<<<(n128 / 4 + 255) / 256, 256, 0, stream>>>((float4*)out, n128 / 4);
  npos_kernel<<<(27 * 128 + 255) / 256, 256, 0, stream>>>(pe, nposW, 27 * 128);
  mlp_kernel<<<(n128 + 255) / 256, 256, 0, stream>>>(pts, w1, s1, sh1, w2, s2, sh2,
                                                     t_h, N);
  gemm_x_kernel<<<gemmBlocks, 256, 0, stream>>>(t_h, feats, w3, b3, x_h, N);
  gemm_qv_kernel<<<gemmBlocks, 256, 0, stream>>>(x_h, wq, bq, wv, bv, nqW, vW, N);
  attn_kernel<<<(M + 15) / 16, 256, 0, stream>>>(kq0, kq1, nqW, nposW, vW, out, M);
  gemm_o_kernel<<<gemmBlocks, 256, 0, stream>>>(out, wo, bo, N);
}
